// IGCNet_85375359910092
// MI455X (gfx1250) — matrix operand order repaired and checked by reading, not compiled, not run on hardware
//
#include <hip/hip_runtime.h>
#include <hip/hip_bf16.h>
#include <math.h>

// ---------------- WMMA types & fragment helpers (gfx1250, wave32) ----------------

typedef __attribute__((ext_vector_type(16))) _Float16 v16h;
typedef __attribute__((ext_vector_type(8)))  float    v8f;

union V16H { v16h v; _Float16 h[16]; unsigned u[8]; };
union H4   { _Float16 h[4]; uint2 u2; };

// B fragment (32x16 K x N) for v_wmma_f32_16x16x32_f16, built from fp32 weights
// stored row-major W[k][ldb] in global memory. k >= K reads as 0 (padding).
// Layout: lane L -> N = n0 + (L & 15); K base = (L>>4)*16; VGPR v holds K = base+2v, base+2v+1.
__device__ __forceinline__ v16h load_b_frag_f32(const float* __restrict__ W,
                                                int ldb, int K, int n0, int kstep) {
    const int lane = threadIdx.x & 31;
    const int n    = n0 + (lane & 15);
    const int kb   = kstep * 32 + ((lane >> 4) << 4);
    V16H b;
#pragma unroll
    for (int v = 0; v < 8; ++v) {
        const int k0 = kb + 2 * v;
        const int k1 = k0 + 1;
        const float f0 = (k0 < K) ? W[(size_t)k0 * ldb + n] : 0.0f;
        const float f1 = (k1 < K) ? W[(size_t)k1 * ldb + n] : 0.0f;
        b.h[2 * v]     = (_Float16)f0;
        b.h[2 * v + 1] = (_Float16)f1;
    }
    return b.v;
}

// A fragment (16x32 M x K) from an LDS tile stored row-major [rows][lda] of halves.
// lda must be even (row byte stride multiple of 16 keeps b128 merging).
// Layout: lane L -> M = m0 + (L & 15), half-group h = L>>4;
// VGPR v : K = kb + h*16 + 2v (contiguous).
__device__ __forceinline__ v16h load_a_frag_lds(const _Float16* __restrict__ A,
                                                int lda, int m0, int kstep) {
    const int lane = threadIdx.x & 31;
    const int m    = m0 + (lane & 15);
    const int hh   = lane >> 4;
    const unsigned* row = (const unsigned*)(A + (size_t)m * lda);
    const int kb = kstep * 32;
    V16H a;
#pragma unroll
    for (int v = 0; v < 4; ++v) a.u[v]     = row[(kb +      hh * 16 + 2 * v) >> 1];
#pragma unroll
    for (int v = 0; v < 4; ++v) a.u[4 + v] = row[(kb + 8 + hh * 16 + 2 * v) >> 1];
    return a.v;
}

__device__ __forceinline__ float sigmoidf_(float x) { return 1.0f / (1.0f + expf(-x)); }

__device__ __forceinline__ uint2 pack_h4(float a, float b, float c, float d) {
    H4 p; p.h[0] = (_Float16)a; p.h[1] = (_Float16)b; p.h[2] = (_Float16)c; p.h[3] = (_Float16)d;
    return p.u2;
}

// D (f32 16x16) element mapping: lane L, vgpr v -> M = v + 8*(L>>4), N = L & 15.

// ---------------- utility kernels ----------------

__global__ void fill_f32_kernel(float* __restrict__ p, float v, long n) {
    long i = (long)blockIdx.x * blockDim.x + threadIdx.x;
    const long stride = (long)gridDim.x * blockDim.x;
    for (; i < n; i += stride) p[i] = v;
}

__global__ void degree_kernel(const int* __restrict__ dst, float* __restrict__ cnt, int E) {
    int e = blockIdx.x * blockDim.x + threadIdx.x;
    if (e < E) atomicAdd(&cnt[dst[e]], 1.0f);
}

// ---------------- input MLP: x[N,2] -> relu(Lin) -> relu(Lin) -> h[N,128] ----------------

#define SH1_LD 136   // 272 B row stride: 17 bank-groups -> conflict-free b128 loads

__global__ __launch_bounds__(256) void input_mlp_kernel(
    const float* __restrict__ x, const float* __restrict__ Win0, const float* __restrict__ bin0,
    const float* __restrict__ Win1, const float* __restrict__ bin1,
    float* __restrict__ h, int N) {
    __shared__ __attribute__((aligned(16))) _Float16 sH1[128 * SH1_LD]; // ~34 KB
    __shared__ float sX[128 * 2];

    const int i0 = blockIdx.x * 128;
    const int t  = threadIdx.x;
    const bool full = (i0 + 128 <= N);

    for (int idx = t; idx < 128 * 2; idx += 256) {
        const int row = idx >> 1, c = idx & 1;
        const int i = i0 + row;
        sX[idx] = (i < N) ? x[(size_t)i * 2 + c] : 0.0f;
    }
    __syncthreads();

    // layer 0 on VALU (K=2)
    for (int idx = t; idx < 128 * 128; idx += 256) {
        const int row = idx >> 7, col = idx & 127;
        float v = sX[row * 2 + 0] * Win0[col] + sX[row * 2 + 1] * Win0[128 + col] + bin0[col];
        sH1[row * SH1_LD + col] = (_Float16)fmaxf(v, 0.0f);
    }
    __syncthreads();

    // layer 1 via WMMA: [128 x 128] x [128 x 128]
    const int w = t >> 5, lane = t & 31;
    const int n0 = w * 16;
    const int n  = n0 + (lane & 15);
    v16h B[4];
#pragma unroll
    for (int ks = 0; ks < 4; ++ks) B[ks] = load_b_frag_f32(Win1, 128, 128, n0, ks);
    const float bias = bin1[n];
    const int hh = lane >> 4;

    for (int r = 0; r < 8; ++r) {
        v8f c;
#pragma unroll
        for (int v = 0; v < 8; ++v) c[v] = bias;
#pragma unroll
        for (int ks = 0; ks < 4; ++ks) {
            v16h a = load_a_frag_lds(sH1, SH1_LD, r * 16, ks);
            c = __builtin_amdgcn_wmma_f32_16x16x32_f16(false, a, false, B[ks], (short)0, c, false, false);
        }
        if (full) {
#pragma unroll
            for (int v = 0; v < 8; ++v) {
                const int m = r * 16 + v + 8 * hh;
                h[(size_t)(i0 + m) * 128 + n] = fmaxf(c[v], 0.0f);
            }
        } else {
#pragma unroll
            for (int v = 0; v < 8; ++v) {
                const int m = r * 16 + v + 8 * hh;
                const int i = i0 + m;
                if (i < N) h[(size_t)i * 128 + n] = fmaxf(c[v], 0.0f);
            }
        }
    }
}

// ---------------- edge message kernel (per layer) ----------------
// m = relu(relu([h[src], ea] @ W1a + b1a) @ W1b + b1b); agg[dst] += m

#define TILE_E  64
#define SAE_LD  168  // 336 B stride: 21 groups, gcd(21,16)=1 -> conflict-free
#define SM_LD   136  // 272 B stride: 17 groups -> conflict-free

__global__ __launch_bounds__(256) void edge_msg_kernel(
    const float* __restrict__ h, const float* __restrict__ ea,
    const int* __restrict__ src, const int* __restrict__ dst,
    const float* __restrict__ W1a, const float* __restrict__ b1a,
    const float* __restrict__ W1b, const float* __restrict__ b1b,
    float* __restrict__ agg, int E) {
    __shared__ __attribute__((aligned(16))) _Float16 sA[TILE_E * SAE_LD]; // ~21 KB
    __shared__ __attribute__((aligned(16))) _Float16 sM[TILE_E * SM_LD]; // ~17 KB
    __shared__ int sSrc[TILE_E];
    __shared__ int sDst[TILE_E];

    const int e0 = blockIdx.x * TILE_E;
    const int t  = threadIdx.x;
    const bool full = (e0 + TILE_E <= E);

    // prefetch next tile's indices / edge attrs (gfx1250 global_prefetch_b8)
    if (e0 + TILE_E < E) {
        __builtin_prefetch(&src[e0 + TILE_E + (t & 63)], 0, 0);
        __builtin_prefetch(&dst[e0 + TILE_E + (t & 63)], 0, 0);
        __builtin_prefetch(&ea[(size_t)(e0 + TILE_E) * 2 + (t & 127)], 0, 0);
    }

    for (int i = t; i < TILE_E; i += 256) {
        const int e = e0 + i;
        const int ec = (e < E) ? e : (E - 1);
        sSrc[i] = src[ec];
        sDst[i] = dst[ec];
    }
    __syncthreads();

    // gather [h[src], edge_attr, 0-pad] -> f16 tile (float4 loads, b64 LDS stores)
    for (int idx = t; idx < TILE_E * 42; idx += 256) {
        const int row = idx / 42, q = idx % 42; // q indexes groups of 4 halves
        const int e = e0 + row;
        float4 val = make_float4(0.f, 0.f, 0.f, 0.f);
        if (e < E && q < 40) {
            if (q < 32)       val = *(const float4*)(h + (size_t)sSrc[row] * 128 + q * 4);
            else if (q == 32) { val.x = ea[(size_t)e * 2]; val.y = ea[(size_t)e * 2 + 1]; }
        }
        ((uint2*)sA)[row * (SAE_LD / 4) + q] = pack_h4(val.x, val.y, val.z, val.w);
    }
    __syncthreads();

    const int w = t >> 5, lane = t & 31;
    const int n0 = w * 16;
    const int n  = n0 + (lane & 15);
    const int hh = lane >> 4;

    v16h B1[5];
#pragma unroll
    for (int ks = 0; ks < 5; ++ks) B1[ks] = load_b_frag_f32(W1a, 128, 130, n0, ks);
    v16h B2[4];
#pragma unroll
    for (int ks = 0; ks < 4; ++ks) B2[ks] = load_b_frag_f32(W1b, 128, 128, n0, ks);
    const float bias1 = b1a[n];
    const float bias2 = b1b[n];

    // GEMM1 + relu -> sM
    for (int r = 0; r < TILE_E / 16; ++r) {
        v8f c;
#pragma unroll
        for (int v = 0; v < 8; ++v) c[v] = bias1;
#pragma unroll
        for (int ks = 0; ks < 5; ++ks) {
            v16h a = load_a_frag_lds(sA, SAE_LD, r * 16, ks);
            c = __builtin_amdgcn_wmma_f32_16x16x32_f16(false, a, false, B1[ks], (short)0, c, false, false);
        }
#pragma unroll
        for (int v = 0; v < 8; ++v) {
            const int m = r * 16 + v + 8 * hh;
            sM[m * SM_LD + n] = (_Float16)fmaxf(c[v], 0.0f);
        }
    }
    __syncthreads();

    // GEMM2 + relu -> atomic scatter to agg[dst]
    for (int r = 0; r < TILE_E / 16; ++r) {
        v8f c;
#pragma unroll
        for (int v = 0; v < 8; ++v) c[v] = bias2;
#pragma unroll
        for (int ks = 0; ks < 4; ++ks) {
            v16h a = load_a_frag_lds(sM, SM_LD, r * 16, ks);
            c = __builtin_amdgcn_wmma_f32_16x16x32_f16(false, a, false, B2[ks], (short)0, c, false, false);
        }
        if (full) {
#pragma unroll
            for (int v = 0; v < 8; ++v) {
                const int m = r * 16 + v + 8 * hh;
                atomicAdd(&agg[(size_t)sDst[m] * 128 + n], fmaxf(c[v], 0.0f));
            }
        } else {
#pragma unroll
            for (int v = 0; v < 8; ++v) {
                const int m = r * 16 + v + 8 * hh;
                if (e0 + m < E) atomicAdd(&agg[(size_t)sDst[m] * 128 + n], fmaxf(c[v], 0.0f));
            }
        }
    }
}

// ---------------- node update kernel (per layer) ----------------
// u = cat(h, agg/denom); out = sigmoid(relu(u@W2a+b2a)@W2b+b2b); r = h + out
// also accumulates per-channel sum / sumsq into stats[0..127] / stats[128..255]

#define TILE_N  64
#define SAN_LD  264  // 528 B stride: 33 groups -> conflict-free

__global__ __launch_bounds__(256) void node_update_kernel(
    const float* __restrict__ h, const float* __restrict__ agg, const float* __restrict__ cnt,
    const float* __restrict__ W2a, const float* __restrict__ b2a,
    const float* __restrict__ W2b, const float* __restrict__ b2b,
    float* __restrict__ rout, float* __restrict__ stats, int N) {
    __shared__ __attribute__((aligned(16))) _Float16 sA[TILE_N * SAN_LD]; // ~33 KB
    __shared__ __attribute__((aligned(16))) _Float16 sM[TILE_N * SM_LD]; // ~17 KB
    __shared__ float sInv[TILE_N];

    const int i0 = blockIdx.x * TILE_N;
    const int t  = threadIdx.x;
    const bool full = (i0 + TILE_N <= N);

    if (t < TILE_N) {
        const int i = i0 + t;
        sInv[t] = (i < N) ? (1.0f / fmaxf(cnt[i], 1.0f)) : 0.0f;
    }
    __syncthreads();

    // stage cat(h, agg/denom) -> f16 tile [64 x 256] (float4 loads, b64 LDS stores)
    for (int idx = t; idx < TILE_N * 66; idx += 256) {
        const int row = idx / 66, q = idx % 66;
        const int i = i0 + row;
        float4 val = make_float4(0.f, 0.f, 0.f, 0.f);
        if (i < N && q < 64) {
            if (q < 32) {
                val = *(const float4*)(h + (size_t)i * 128 + q * 4);
            } else {
                const float4 a = *(const float4*)(agg + (size_t)i * 128 + (q - 32) * 4);
                const float s = sInv[row];
                val.x = a.x * s; val.y = a.y * s; val.z = a.z * s; val.w = a.w * s;
            }
        }
        ((uint2*)sA)[row * (SAN_LD / 4) + q] = pack_h4(val.x, val.y, val.z, val.w);
    }
    __syncthreads();

    const int w = t >> 5, lane = t & 31;
    const int n0 = w * 16;
    const int n  = n0 + (lane & 15);
    const int hh = lane >> 4;

    v16h B1[8];
#pragma unroll
    for (int ks = 0; ks < 8; ++ks) B1[ks] = load_b_frag_f32(W2a, 128, 256, n0, ks);
    v16h B2[4];
#pragma unroll
    for (int ks = 0; ks < 4; ++ks) B2[ks] = load_b_frag_f32(W2b, 128, 128, n0, ks);
    const float bias1 = b2a[n];
    const float bias2 = b2b[n];

    // GEMM1 + relu -> sM
    for (int r = 0; r < TILE_N / 16; ++r) {
        v8f c;
#pragma unroll
        for (int v = 0; v < 8; ++v) c[v] = bias1;
#pragma unroll
        for (int ks = 0; ks < 8; ++ks) {
            v16h a = load_a_frag_lds(sA, SAN_LD, r * 16, ks);
            c = __builtin_amdgcn_wmma_f32_16x16x32_f16(false, a, false, B1[ks], (short)0, c, false, false);
        }
#pragma unroll
        for (int v = 0; v < 8; ++v) {
            const int m = r * 16 + v + 8 * hh;
            sM[m * SM_LD + n] = (_Float16)fmaxf(c[v], 0.0f);
        }
    }
    __syncthreads();

    // GEMM2 + sigmoid + residual; accumulate channel stats
    float psum = 0.0f, psq = 0.0f;
    for (int r = 0; r < TILE_N / 16; ++r) {
        v8f c;
#pragma unroll
        for (int ks = 0; ks < 4; ++ks) {
            if (ks == 0) {
#pragma unroll
                for (int v = 0; v < 8; ++v) c[v] = bias2;
            }
            v16h a = load_a_frag_lds(sM, SM_LD, r * 16, ks);
            c = __builtin_amdgcn_wmma_f32_16x16x32_f16(false, a, false, B2[ks], (short)0, c, false, false);
        }
        if (full) {
#pragma unroll
            for (int v = 0; v < 8; ++v) {
                const int m = r * 16 + v + 8 * hh;
                const int i = i0 + m;
                const float rv = h[(size_t)i * 128 + n] + sigmoidf_(c[v]);
                rout[(size_t)i * 128 + n] = rv;
                psum += rv;
                psq  += rv * rv;
            }
        } else {
#pragma unroll
            for (int v = 0; v < 8; ++v) {
                const int m = r * 16 + v + 8 * hh;
                const int i = i0 + m;
                if (i < N) {
                    const float rv = h[(size_t)i * 128 + n] + sigmoidf_(c[v]);
                    rout[(size_t)i * 128 + n] = rv;
                    psum += rv;
                    psq  += rv * rv;
                }
            }
        }
    }
    // lanes L and L+16 hold the same channel n -> combine, then 16 atomics per wave
    psum += __shfl_xor(psum, 16, 32);
    psq  += __shfl_xor(psq, 16, 32);
    if (lane < 16) {
        atomicAdd(&stats[n], psum);
        atomicAdd(&stats[128 + n], psq);
    }
}

// ---------------- BatchNorm finalize + apply ----------------

__global__ void bn_finalize_kernel(const float* __restrict__ stats,
                                   const float* __restrict__ gamma, const float* __restrict__ beta,
                                   float* __restrict__ ss, int N) {
    const int n = threadIdx.x;
    if (n < 128) {
        const float invN = 1.0f / (float)N;
        const float mu  = stats[n] * invN;
        const float var = stats[128 + n] * invN - mu * mu;
        const float inv = rsqrtf(var + 1e-5f);
        const float sc  = gamma[n] * inv;
        ss[n]       = sc;
        ss[128 + n] = beta[n] - mu * sc;
    }
}

__global__ void bn_apply_kernel(const float* __restrict__ r, const float* __restrict__ ss,
                                float* __restrict__ h, long n) {
    long i = (long)blockIdx.x * blockDim.x + threadIdx.x;
    const long stride = (long)gridDim.x * blockDim.x;
    for (; i < n; i += stride) {
        const int ch = (int)(i & 127);
        h[i] = r[i] * ss[ch] + ss[128 + ch];
    }
}

// ---------------- final MLP: f = relu(h@Wf+bf); out = sigmoid(f@Wo+bo) ----------------

__global__ __launch_bounds__(256) void final_mlp_kernel(
    const float* __restrict__ h, const float* __restrict__ Wf, const float* __restrict__ bf,
    const float* __restrict__ Wo, const float* __restrict__ bo,
    float* __restrict__ out, int N) {
    __shared__ __attribute__((aligned(16))) _Float16 sA[TILE_N * SM_LD]; // ~17 KB
    __shared__ __attribute__((aligned(16))) _Float16 sF[TILE_N * SM_LD]; // ~17 KB

    const int i0 = blockIdx.x * TILE_N;
    const int t  = threadIdx.x;

    for (int idx = t; idx < TILE_N * 34; idx += 256) {
        const int row = idx / 34, q = idx % 34;
        const int i = i0 + row;
        float4 val = make_float4(0.f, 0.f, 0.f, 0.f);
        if (i < N && q < 32) val = *(const float4*)(h + (size_t)i * 128 + q * 4);
        ((uint2*)sA)[row * (SM_LD / 4) + q] = pack_h4(val.x, val.y, val.z, val.w);
    }
    __syncthreads();

    const int w = t >> 5, lane = t & 31;
    const int n0 = w * 16;
    const int n  = n0 + (lane & 15);
    const int hh = lane >> 4;

    v16h B[4];
#pragma unroll
    for (int ks = 0; ks < 4; ++ks) B[ks] = load_b_frag_f32(Wf, 128, 128, n0, ks);
    const float bias = bf[n];

    for (int r = 0; r < TILE_N / 16; ++r) {
        v8f c;
#pragma unroll
        for (int v = 0; v < 8; ++v) c[v] = bias;
#pragma unroll
        for (int ks = 0; ks < 4; ++ks) {
            v16h a = load_a_frag_lds(sA, SM_LD, r * 16, ks);
            c = __builtin_amdgcn_wmma_f32_16x16x32_f16(false, a, false, B[ks], (short)0, c, false, false);
        }
#pragma unroll
        for (int v = 0; v < 8; ++v) {
            const int m = r * 16 + v + 8 * hh;
            sF[m * SM_LD + n] = (_Float16)fmaxf(c[v], 0.0f);
        }
    }
    __syncthreads();

    // output projection H -> 1 (tiny, VALU)
    for (int row = t; row < TILE_N; row += 256) {
        const int i = i0 + row;
        if (i < N) {
            float s = bo[0];
            for (int k = 0; k < 128; ++k) s += (float)sF[row * SM_LD + k] * Wo[k];
            out[i] = sigmoidf_(s);
        }
    }
}

// ---------------- host orchestration ----------------

extern "C" void kernel_launch(void* const* d_in, const int* in_sizes, int n_in,
                              void* d_out, int out_size, void* d_ws, size_t ws_size,
                              hipStream_t stream) {
    const float* x         = (const float*)d_in[0];
    const float* edge_attr = (const float*)d_in[1];
    const int*   edge_index= (const int*)d_in[2];
    // d_in[3] = batch (unused)
    const float* Win0 = (const float*)d_in[4];
    const float* bin0 = (const float*)d_in[5];
    const float* Win1 = (const float*)d_in[6];
    const float* bin1 = (const float*)d_in[7];
    const float* W1a  = (const float*)d_in[8];
    const float* b1a  = (const float*)d_in[9];
    const float* W1b  = (const float*)d_in[10];
    const float* b1b  = (const float*)d_in[11];
    const float* W2a  = (const float*)d_in[12];
    const float* b2a  = (const float*)d_in[13];
    const float* W2b  = (const float*)d_in[14];
    const float* b2b  = (const float*)d_in[15];
    const float* gamma= (const float*)d_in[16];
    const float* beta = (const float*)d_in[17];
    const float* Wf   = (const float*)d_in[18];
    const float* bf   = (const float*)d_in[19];
    const float* Wo   = (const float*)d_in[20];
    const float* bo   = (const float*)d_in[21];

    const int N = in_sizes[0] / 2;  // NODE_IN = 2
    const int E = in_sizes[1] / 2;  // EDGE_IN = 2
    const int L = 3;
    const int* src = edge_index;
    const int* dst = edge_index + E;

    char* ws = (char*)d_ws;
    float* h    = (float*)ws; ws += (size_t)N * 128 * sizeof(float);
    float* r    = (float*)ws; ws += (size_t)N * 128 * sizeof(float);
    float* agg  = (float*)ws; ws += (size_t)N * 128 * sizeof(float);
    float* cnt  = (float*)ws; ws += (size_t)N * sizeof(float);
    float* stats= (float*)ws; ws += 256 * sizeof(float);
    float* ss   = (float*)ws; ws += 256 * sizeof(float);

    const long NH = (long)N * 128;

    // degree counts (mean-aggregation denominator)
    fill_f32_kernel<<<256, 256, 0, stream>>>(cnt, 0.0f, (long)N);
    degree_kernel<<<(E + 255) / 256, 256, 0, stream>>>(dst, cnt, E);

    // input MLP -> h
    input_mlp_kernel<<<(N + 127) / 128, 256, 0, stream>>>(x, Win0, bin0, Win1, bin1, h, N);

    for (int l = 0; l < L; ++l) {
        fill_f32_kernel<<<2048, 256, 0, stream>>>(agg, 0.0f, NH);
        edge_msg_kernel<<<(E + TILE_E - 1) / TILE_E, 256, 0, stream>>>(
            h, edge_attr, src, dst,
            W1a + (size_t)l * 130 * 128, b1a + (size_t)l * 128,
            W1b + (size_t)l * 128 * 128, b1b + (size_t)l * 128,
            agg, E);
        fill_f32_kernel<<<1, 256, 0, stream>>>(stats, 0.0f, 256);
        node_update_kernel<<<(N + TILE_N - 1) / TILE_N, 256, 0, stream>>>(
            h, agg, cnt,
            W2a + (size_t)l * 256 * 128, b2a + (size_t)l * 128,
            W2b + (size_t)l * 128 * 128, b2b + (size_t)l * 128,
            r, stats, N);
        bn_finalize_kernel<<<1, 128, 0, stream>>>(stats, gamma + (size_t)l * 128,
                                                  beta + (size_t)l * 128, ss, N);
        bn_apply_kernel<<<2048, 256, 0, stream>>>(r, ss, h, NH);
    }

    final_mlp_kernel<<<(N + TILE_N - 1) / TILE_N, 256, 0, stream>>>(
        h, Wf, bf, Wo, bo, (float*)d_out, N);
}
